// AttentionLayer_66949950210666
// MI455X (gfx1250) — matrix operand order repaired and checked by reading, not compiled, not run on hardware
//
#include <hip/hip_runtime.h>
#include <hip/hip_bf16.h>

// Problem constants (match reference)
#define B_  2
#define S_  2048
#define D_  1024
#define H_  16
#define HD_ 64

typedef __attribute__((ext_vector_type(16))) __bf16 v16bf;
typedef __attribute__((ext_vector_type(8)))  float  v8f;
typedef __attribute__((ext_vector_type(4)))  unsigned u32x4;
typedef __attribute__((ext_vector_type(4)))  float    f32x4;

struct U8x2 { u32x4 lo, hi; };          // 32 bytes = 16 bf16
union FragU { U8x2 u; v16bf v; };       // both members trivial

__device__ __forceinline__ __bf16 f2bf(float f) {
    unsigned u = __builtin_bit_cast(unsigned, f);
    unsigned r = (u + 0x7FFFu + ((u >> 16) & 1u)) >> 16;   // round-to-nearest-even
    unsigned short s = (unsigned short)r;
    return __builtin_bit_cast(__bf16, s);
}

__device__ __forceinline__ unsigned pack2bf(float a, float b) {
    unsigned ua = __builtin_bit_cast(unsigned, a);
    unsigned ub = __builtin_bit_cast(unsigned, b);
    unsigned ra = (ua + 0x7FFFu + ((ua >> 16) & 1u)) >> 16;
    unsigned rb = (ub + 0x7FFFu + ((ub >> 16) & 1u)) >> 16;
    return ra | (rb << 16);
}

__device__ __forceinline__ v8f wmma_bf16(v16bf a, v16bf b, v8f c) {
    return __builtin_amdgcn_wmma_f32_16x16x32_bf16(
        /*neg_a=*/false, a, /*neg_b=*/false, b,
        /*c_mod=*/(short)0, c, /*reuse_a=*/false, /*reuse_b=*/false);
}

// ds_swizzle xor-mode shuffle: new_lane = lane ^ M (and_mask=0x1f, or=0)
#define SWIZ_XOR(v, M) \
    __builtin_bit_cast(float, __builtin_amdgcn_ds_swizzle( \
        __builtin_bit_cast(int, (v)), (0x1f | ((M) << 10))))

// ---------------------------------------------------------------------------
// Packed f32 -> bf16 conversion: 8 elements/thread, b128 in/out.
// ---------------------------------------------------------------------------
__global__ __launch_bounds__(256)
void cvt_f32_to_bf16(const float* __restrict__ src, __bf16* __restrict__ dst) {
    size_t i = ((size_t)blockIdx.x * 256 + threadIdx.x) * 8;
    f32x4 a = *(const f32x4*)(src + i);
    f32x4 b = *(const f32x4*)(src + i + 4);
    u32x4 o;
    o[0] = pack2bf(a[0], a[1]);
    o[1] = pack2bf(a[2], a[3]);
    o[2] = pack2bf(b[0], b[1]);
    o[3] = pack2bf(b[2], b[3]);
    *(u32x4*)(dst + i) = o;
}

// ---------------------------------------------------------------------------
// GEMM:  Out[M,N] = (A[M,K] @ W[N,K]^T + bias[N]) * outScale
// A and W are bf16 (pre-converted); tiles staged via async-to-LDS copies.
// OUTMODE: 0 = bf16 [M,N], 1 = bf16 transposed [b][feature][s], 2 = f32 [M,N].
// Block tile 128x128, BK=32, 256 threads = 8 waves (2x4);
// each wave computes a 64x32 patch = 8 WMMAs per k-step.
// ---------------------------------------------------------------------------
template<int OUTMODE>
__global__ __launch_bounds__(256)
void gemm_bias_wmma(const __bf16* __restrict__ A,
                    const __bf16* __restrict__ Wb,
                    const float* __restrict__ bias,
                    void* __restrict__ Outp,
                    int M, int N, int K, float outScale) {
    constexpr int BM = 128, BN = 128, BK = 32;
    __shared__ __bf16 As[BM * BK];
    __shared__ __bf16 Bs[BN * BK];

    const int t    = threadIdx.x;
    const int lane = t & 31;
    const int wave = t >> 5;
    const int wm   = wave >> 2;          // 0..1  (M sub-block of 64)
    const int wn   = wave & 3;           // 0..3  (N sub-block of 32)
    const int lh   = lane & 15;
    const int hi   = lane >> 4;          // 0 / 1 (half-wave)

    const int rowBase = blockIdx.y * BM;
    const int colBase = blockIdx.x * BN;

    // Async staging: each tile is 128 rows x 64 B; 2 threads/row, 32 B each.
    const int sr = t >> 1, sc = (t & 1) * 16;
    const unsigned aLds = (unsigned)(uintptr_t)&As[sr * BK + sc];
    const unsigned bLds = (unsigned)(uintptr_t)&Bs[sr * BK + sc];
    const __bf16* ag = A  + (size_t)(rowBase + sr) * K + sc;
    const __bf16* bg = Wb + (size_t)(colBase + sr) * K + sc;

    v8f zero8 = {};
    v8f acc[4][2];
#pragma unroll
    for (int mi = 0; mi < 4; ++mi)
#pragma unroll
        for (int ni = 0; ni < 2; ++ni) acc[mi][ni] = zero8;

    const int koffA = hi ? 8 : 0;        // A: lanes>=16 hold K 8-15 / 24-31
    const int koffB = hi ? 8 : 0;        // B: lanes>=16 hold K 8-15 / 24-31

    for (int k0 = 0; k0 < K; k0 += BK, ag += BK, bg += BK) {
        // --- stage both tiles with gfx1250 async-to-LDS copies ---
        asm volatile(
            "global_load_async_to_lds_b128 %0, %2, off\n\t"
            "global_load_async_to_lds_b128 %0, %2, off offset:16\n\t"
            "global_load_async_to_lds_b128 %1, %3, off\n\t"
            "global_load_async_to_lds_b128 %1, %3, off offset:16"
            :: "v"(aLds), "v"(bLds), "v"(ag), "v"(bg) : "memory");
        asm volatile("s_wait_asynccnt 0x0" ::: "memory");
        __syncthreads();

        // --- assemble fragments (ISA 16-bit A 16x32 / B 32x16 layouts) ---
        v16bf af[4], bfr[2];
#pragma unroll
        for (int mi = 0; mi < 4; ++mi) {
            int arow = wm * 64 + mi * 16 + lh;
            FragU f;
            f.u.lo = *(const u32x4*)&As[arow * BK + koffA];
            f.u.hi = *(const u32x4*)&As[arow * BK + koffA + 16];
            af[mi] = f.v;
        }
#pragma unroll
        for (int ni = 0; ni < 2; ++ni) {
            int brow = wn * 32 + ni * 16 + lh;   // output column n
            FragU f;
            f.u.lo = *(const u32x4*)&Bs[brow * BK + koffB];
            f.u.hi = *(const u32x4*)&Bs[brow * BK + koffB + 16];
            bfr[ni] = f.v;
        }
#pragma unroll
        for (int mi = 0; mi < 4; ++mi)
#pragma unroll
            for (int ni = 0; ni < 2; ++ni)
                acc[mi][ni] = wmma_bf16(af[mi], bfr[ni], acc[mi][ni]);
        __syncthreads();
    }

    // --- epilogue: C/D layout lane 0-15 -> M=i, lane 16-31 -> M=i+8 ---
#pragma unroll
    for (int ni = 0; ni < 2; ++ni) {
        int col = colBase + wn * 32 + ni * 16 + lh;
        float bv = bias[col];
#pragma unroll
        for (int mi = 0; mi < 4; ++mi) {
#pragma unroll
            for (int i = 0; i < 8; ++i) {
                int grow = rowBase + wm * 64 + mi * 16 + (hi ? i + 8 : i);
                float v = (acc[mi][ni][i] + bv) * outScale;
                if (OUTMODE == 1) {
                    int bb = grow >> 11;            // grow / S_
                    int ss = grow & (S_ - 1);       // grow % S_
                    ((__bf16*)Outp)[((size_t)(bb * N + col)) * S_ + ss] = f2bf(v);
                } else if (OUTMODE == 0) {
                    ((__bf16*)Outp)[(size_t)grow * N + col] = f2bf(v);
                } else {
                    ((float*)Outp)[(size_t)grow * N + col] = v;
                }
            }
        }
    }
}

// ---------------------------------------------------------------------------
// Flash attention: one wave per 16-query tile per (b, h).
// Q is pre-scaled by 0.125*log2(e) (folded into the Q GEMM epilogue), so the
// softmax runs in the exp2 domain with no per-score multiplies.
// Q/K stored bf16 [B*S, D]; Vt stored bf16 transposed [b][feature][s].
// K chunks staged to LDS with gfx1250 async-to-LDS copies (ASYNCcnt).
// ---------------------------------------------------------------------------
__global__ __launch_bounds__(128)
void attn_wmma(const __bf16* __restrict__ Q,
               const __bf16* __restrict__ Kt,
               const __bf16* __restrict__ Vt,
               __bf16* __restrict__ Y) {
    __shared__ __bf16 Plds[4][16 * 32];   // per-wave P tile  (16 q x 32 keys)
    __shared__ __bf16 Klds[4][32 * 64];   // per-wave K chunk (32 keys x 64 d)

    const int t    = threadIdx.x;
    const int lane = t & 31;
    const int wave = t >> 5;
    const int lh   = lane & 15;
    const int hi   = lane >> 4;

    const int bh = blockIdx.x;
    const int b  = bh / H_;
    const int h  = bh % H_;
    const int qtile = blockIdx.y * 4 + wave;

    const int koffA = hi ? 8 : 0;        // A-operand K offsets per half-wave
    const int koffB = hi ? 8 : 0;        // B-operand K offsets per half-wave

    // Q fragments: A-layout, loaded once (head dim 64 = 2 K-chunks of 32)
    const __bf16* qrow = Q + ((size_t)(b * S_ + qtile * 16 + lh)) * D_ + h * HD_;
    v16bf qf[2];
#pragma unroll
    for (int c = 0; c < 2; ++c) {
        FragU f;
        f.u.lo = *(const u32x4*)(qrow + c * 32 + koffA);
        f.u.hi = *(const u32x4*)(qrow + c * 32 + koffA + 16);
        qf[c] = f.v;
    }

    float mrow[8], lrow[8];
    v8f zero8 = {};
    v8f oacc[4];
#pragma unroll
    for (int i = 0; i < 8; ++i) { mrow[i] = -1e30f; lrow[i] = 0.f; }
#pragma unroll
    for (int nt = 0; nt < 4; ++nt) oacc[nt] = zero8;

    // Per-lane async staging addresses: lane copies key-row `lane` (128 B).
    const unsigned kldsBase = (unsigned)(uintptr_t)&Klds[wave][lane * 64];

    for (int kc = 0; kc < S_ / 32; ++kc) {
        // --- stage K chunk into LDS via async copies ---
        {
            const __bf16* ksrc =
                Kt + ((size_t)(b * S_ + kc * 32 + lane)) * D_ + h * HD_;
            asm volatile("s_wait_dscnt 0x0" ::: "memory");
            asm volatile(
                "global_load_async_to_lds_b128 %0, %1, off\n\t"
                "global_load_async_to_lds_b128 %0, %1, off offset:16\n\t"
                "global_load_async_to_lds_b128 %0, %1, off offset:32\n\t"
                "global_load_async_to_lds_b128 %0, %1, off offset:48\n\t"
                "global_load_async_to_lds_b128 %0, %1, off offset:64\n\t"
                "global_load_async_to_lds_b128 %0, %1, off offset:80\n\t"
                "global_load_async_to_lds_b128 %0, %1, off offset:96\n\t"
                "global_load_async_to_lds_b128 %0, %1, off offset:112"
                :: "v"(kldsBase), "v"(ksrc) : "memory");
            asm volatile("s_wait_asynccnt 0x0" ::: "memory");
        }

        // --- scores (log2 domain): two 16-key tiles, two WMMAs each ---
        v8f sc0 = {}, sc1 = {};
#pragma unroll
        for (int j = 0; j < 2; ++j) {
            const int krow = (j * 16 + lh) * 64;   // local key row in Klds
#pragma unroll
            for (int c = 0; c < 2; ++c) {
                FragU f;
                f.u.lo = *(const u32x4*)&Klds[wave][krow + c * 32 + koffB];
                f.u.hi = *(const u32x4*)&Klds[wave][krow + c * 32 + koffB + 16];
                if (j == 0) sc0 = wmma_bf16(qf[c], f.v, sc0);
                else        sc1 = wmma_bf16(qf[c], f.v, sc1);
            }
        }

        // --- online softmax (exp2 domain) over the 32 keys of this chunk ---
        float p0[8], p1[8];
#pragma unroll
        for (int i = 0; i < 8; ++i) {
            float s0 = sc0[i], s1 = sc1[i];
            float r = fmaxf(s0, s1);
            r = fmaxf(r, SWIZ_XOR(r, 1));
            r = fmaxf(r, SWIZ_XOR(r, 2));
            r = fmaxf(r, SWIZ_XOR(r, 4));
            r = fmaxf(r, SWIZ_XOR(r, 8));    // stays within 16-lane half
            float mn = fmaxf(mrow[i], r);
            float alpha = exp2f(mrow[i] - mn);
            mrow[i] = mn;
            float e0 = exp2f(s0 - mn), e1 = exp2f(s1 - mn);
            float rs = e0 + e1;
            rs += SWIZ_XOR(rs, 1);
            rs += SWIZ_XOR(rs, 2);
            rs += SWIZ_XOR(rs, 4);
            rs += SWIZ_XOR(rs, 8);
            lrow[i] = lrow[i] * alpha + rs;
            p0[i] = e0; p1[i] = e1;
#pragma unroll
            for (int nt = 0; nt < 4; ++nt) oacc[nt][i] *= alpha;
        }

        // --- write P (C layout -> row major) and reload in A layout.
        //     Intra-wave only: DS ops are in-order, no barrier needed. ---
#pragma unroll
        for (int i = 0; i < 8; ++i) {
            int row = hi ? i + 8 : i;
            Plds[wave][row * 32 + lh]      = f2bf(p0[i]);
            Plds[wave][row * 32 + 16 + lh] = f2bf(p1[i]);
        }
        v16bf pf;
        {
            FragU f;
            f.u.lo = *(const u32x4*)&Plds[wave][lh * 32 + koffA];
            f.u.hi = *(const u32x4*)&Plds[wave][lh * 32 + koffA + 16];
            pf = f.v;
        }

        // --- PV: 4 output d-tiles of 16; Vt is pre-transposed so each lane
        //     reads two 8-bf16 runs 16 apart (B-operand layout) from global ---
#pragma unroll
        for (int nt = 0; nt < 4; ++nt) {
            const __bf16* vr = Vt +
                ((size_t)(b * D_ + h * HD_ + nt * 16 + lh)) * S_ + kc * 32 + koffB;
            FragU f;
            f.u.lo = *(const u32x4*)(vr);
            f.u.hi = *(const u32x4*)(vr + 16);
            oacc[nt] = wmma_bf16(pf, f.v, oacc[nt]);
        }
    }

    // --- normalize and store Y (bf16) ---
#pragma unroll
    for (int i = 0; i < 8; ++i) {
        int mr = hi ? i + 8 : i;
        float inv = 1.0f / lrow[i];
        __bf16* yr = Y + ((size_t)(b * S_ + qtile * 16 + mr)) * D_ + h * HD_;
#pragma unroll
        for (int nt = 0; nt < 4; ++nt)
            yr[nt * 16 + lh] = f2bf(oacc[nt][i] * inv);
    }
}

// ---------------------------------------------------------------------------
extern "C" void kernel_launch(void* const* d_in, const int* in_sizes, int n_in,
                              void* d_out, int out_size, void* d_ws, size_t ws_size,
                              hipStream_t stream) {
    (void)in_sizes; (void)n_in; (void)out_size; (void)ws_size;
    const float* dec = (const float*)d_in[0];
    const float* enc = (const float*)d_in[1];
    const float* Wq  = (const float*)d_in[2];
    const float* bq  = (const float*)d_in[3];
    const float* Wk  = (const float*)d_in[4];
    const float* bk  = (const float*)d_in[5];
    const float* Wv  = (const float*)d_in[6];
    const float* bv  = (const float*)d_in[7];
    const float* Wp  = (const float*)d_in[8];
    const float* bp  = (const float*)d_in[9];
    float* out = (float*)d_out;

    const int M = B_ * S_;                                    // 4096
    const size_t tokBytes = (size_t)M * D_ * sizeof(__bf16);  // 8 MB
    const size_t wBytes   = (size_t)D_ * D_ * sizeof(__bf16); // 2 MB
    char* ws = (char*)d_ws;
    __bf16* Qb    = (__bf16*)(ws);
    __bf16* Kb    = (__bf16*)(ws + tokBytes);
    __bf16* Vtb   = (__bf16*)(ws + 2 * tokBytes);  // transposed [b][feat][s]
    __bf16* decbf = (__bf16*)(ws + 3 * tokBytes);  // reused as Yb after Q GEMM
    __bf16* Yb    = decbf;
    __bf16* encbf = (__bf16*)(ws + 4 * tokBytes);
    __bf16* Wqbf  = (__bf16*)(ws + 5 * tokBytes);
    __bf16* Wkbf  = (__bf16*)(ws + 5 * tokBytes + wBytes);
    __bf16* Wvbf  = (__bf16*)(ws + 5 * tokBytes + 2 * wBytes);
    __bf16* Wpbf  = (__bf16*)(ws + 5 * tokBytes + 3 * wBytes);

    // One-shot packed conversions (activations 4M elems, weights 1M elems).
    const int actBlocks = (M * D_) / (256 * 8);   // 2048
    const int wBlocks   = (D_ * D_) / (256 * 8);  // 512
    cvt_f32_to_bf16<<<actBlocks, 256, 0, stream>>>(dec, decbf);
    cvt_f32_to_bf16<<<actBlocks, 256, 0, stream>>>(enc, encbf);
    cvt_f32_to_bf16<<<wBlocks,   256, 0, stream>>>(Wq, Wqbf);
    cvt_f32_to_bf16<<<wBlocks,   256, 0, stream>>>(Wk, Wkbf);
    cvt_f32_to_bf16<<<wBlocks,   256, 0, stream>>>(Wv, Wvbf);
    cvt_f32_to_bf16<<<wBlocks,   256, 0, stream>>>(Wp, Wpbf);

    const float qScale = 0.125f * 1.44269504088896f;  // 1/sqrt(64) * log2(e)

    dim3 gGemm(D_ / 128, M / 128);  // (8, 32)
    gemm_bias_wmma<0><<<gGemm, 256, 0, stream>>>(decbf, Wqbf, bq, Qb,  M, D_, D_, qScale);
    gemm_bias_wmma<0><<<gGemm, 256, 0, stream>>>(encbf, Wkbf, bk, Kb,  M, D_, D_, 1.0f);
    gemm_bias_wmma<1><<<gGemm, 256, 0, stream>>>(encbf, Wvbf, bv, Vtb, M, D_, D_, 1.0f);

    dim3 gAttn(B_ * H_, S_ / 64);   // (32, 32): 4 q-tiles of 16 per block
    attn_wmma<<<gAttn, 128, 0, stream>>>(Qb, Kb, Vtb, Yb);

    gemm_bias_wmma<2><<<gGemm, 256, 0, stream>>>(Yb, Wpbf, bp, out, M, D_, D_, 1.0f);
}
